// GATModel_32762010534258
// MI455X (gfx1250) — matrix operand order repaired and checked by reading, not compiled, not run on hardware
//
#include <hip/hip_runtime.h>
#include <hip/hip_bf16.h>

typedef _Float16 half_t;
typedef __attribute__((ext_vector_type(16))) _Float16 v16h;
typedef __attribute__((ext_vector_type(8)))  float    v8f;

#define WMMA_F16(a, b, c) \
  __builtin_amdgcn_wmma_f32_16x16x32_f16(false, (a), false, (b), (short)0, (c), false, false)

// fast hardware transcendentals: v_rcp_f32 / v_exp_f32 (approx, ~1ulp)
__device__ __forceinline__ float frcp_(float v)     { return __builtin_amdgcn_rcpf(v); }
__device__ __forceinline__ float sigmoidf_(float v) { return frcp_(1.0f + __expf(-v)); }
__device__ __forceinline__ float tanhf_(float v)    { return 2.0f * frcp_(1.0f + __expf(-2.0f * v)) - 1.0f; }
__device__ __forceinline__ float lrelu_(float v)    { return v > 0.0f ? v : 0.01f * v; }

// ---- WMMA fragment helpers (A/B: contiguous k order, matched; C/D per CDNA5 ISA 7.12.2) ------------------
// A: 16x32 f16. lane: M = lane%16, half = lane/16.
//    element e -> K = kBase + half*16 + e
//    -> per lane: one contiguous 16-half (32B) run
__device__ __forceinline__ v16h load_A_lds_h(const half_t* A, int lda, int mBase, int kBase, int lane) {
  const int m  = mBase + (lane & 15);
  const int hh = (lane >> 4) << 4;
  v16h a;
#pragma unroll
  for (int e = 0; e < 16; ++e) {
    const int k = kBase + hh + e;
    a[e] = A[m * lda + k];
  }
  return a;
}
__device__ __forceinline__ v16h load_A_lds_f32(const float* A, int lda, int mBase, int kBase, int lane) {
  const int m  = mBase + (lane & 15);
  const int hh = (lane >> 4) << 4;
  v16h a;
#pragma unroll
  for (int e = 0; e < 16; ++e) {
    const int k = kBase + hh + e;
    a[e] = (half_t)A[m * lda + k];
  }
  return a;
}
// A from global f16 with arbitrary row stride (rows = batch, cols = K)
__device__ __forceinline__ v16h load_A_g16s(const half_t* __restrict__ A, int rowStride,
                                            int mBase, int kBase, int lane) {
  const int m  = mBase + (lane & 15);
  const int hh = (lane >> 4) << 4;
  v16h a;
#pragma unroll
  for (int e = 0; e < 16; ++e) {
    const int k = kBase + hh + e;
    a[e] = A[(size_t)m * rowStride + k];
  }
  return a;
}
// B: 32x16 f16, row-major [K][ldb]. lane: N = lane%16, half = lane/16.
//    element e -> K = kBase + half*16 + e
__device__ __forceinline__ v16h load_B_g16(const half_t* __restrict__ B, int ldb,
                                           int kBase, int nBase, int lane) {
  const int n  = nBase + (lane & 15);
  const int hh = (lane >> 4) << 4;
  v16h b;
#pragma unroll
  for (int e = 0; e < 16; ++e) b[e] = B[(size_t)(kBase + hh + e) * ldb + n];
  return b;
}
// C/D: 16x16 f32. VGPR v, lanes 0-15 -> M=v ; lanes 16-31 -> M=v+8 ; N=lane%16
__device__ __forceinline__ void store_C_lds(float* C, int ldc, int mBase, int nBase, int lane, v8f c) {
  const int n  = nBase + (lane & 15);
  const int hh = lane >> 4;
#pragma unroll
  for (int v = 0; v < 8; ++v) C[(mBase + v + 8 * hh) * ldc + n] = c[v];
}

// ---- weight transpose + f32->f16 cast:  dst[C][R] = src[R][C] --------------
__global__ void k_tcast(const float* __restrict__ src, half_t* __restrict__ dst, int R, int C) {
  const int idx = blockIdx.x * blockDim.x + threadIdx.x;
  if (idx < R * C) {
    const int r = idx / C, c = idx % C;
    dst[(size_t)c * R + r] = (half_t)src[idx];
  }
}

// ============================================================================
// Fused 2-layer GRU encoder over T=60.  32 batch rows / block, 6 waves.
// Hidden state: f32 master + f16 shadow (direct WMMA-A source, no cvts).
// Compact staged biases: [0:128)=bih+bhh (r,z), [128:192)=bih_n, [192:256)=bhh_n
// ============================================================================
__global__ __launch_bounds__(192) void k_encoder(
    const float* __restrict__ x,                               // [N][60][6]
    const float* __restrict__ Wih0,                            // [192][6]
    const float* __restrict__ bih0, const float* __restrict__ bhh0,
    const float* __restrict__ bih1, const float* __restrict__ bhh1,
    const half_t* __restrict__ Whh0T,                          // [64][192] f16
    const half_t* __restrict__ Wih1T, const half_t* __restrict__ Whh1T,
    float* __restrict__ enc, half_t* __restrict__ enc16)       // [N][64]
{
  __shared__ float  sH0[32 * 64];
  __shared__ float  sH1[32 * 64];
  __shared__ half_t sH0h[32 * 64];
  __shared__ half_t sH1h[32 * 64];
  __shared__ float  sG[32 * 192];
  __shared__ float  sGn[32 * 64];
  __shared__ float  sXt[32 * 6];
  __shared__ float  sW0[192 * 6];
  __shared__ float  sB0[256];
  __shared__ float  sB1[256];

  const int tid = threadIdx.x;
  const int lane = tid & 31;
  const int wave = tid >> 5;                 // 0..5, owns N-tiles {2w, 2w+1}
  const int rowBase = blockIdx.x * 32;

  for (int i = tid; i < 32 * 64; i += 192) {
    sH0[i] = 0.0f; sH1[i] = 0.0f;
    sH0h[i] = (half_t)0.0f; sH1h[i] = (half_t)0.0f;
  }
  for (int i = tid; i < 192 * 6; i += 192) sW0[i] = Wih0[i];
  if (tid < 128) { sB0[tid] = bih0[tid] + bhh0[tid]; sB1[tid] = bih1[tid] + bhh1[tid]; }
  else if (tid < 192) {
    const int j = tid - 128;
    sB0[128 + j] = bih0[128 + j]; sB0[192 + j] = bhh0[128 + j];
    sB1[128 + j] = bih1[128 + j]; sB1[192 + j] = bhh1[128 + j];
  }

  // Preload all B fragments (weights stationary across the whole t-loop)
  v16h bW0[2][2], bI1[2][2], bW1[2][2];      // [nt][kt]
#pragma unroll
  for (int nt = 0; nt < 2; ++nt)
#pragma unroll
    for (int kt = 0; kt < 2; ++kt) {
      const int nb = (2 * wave + nt) * 16;
      bW0[nt][kt] = load_B_g16(Whh0T, 192, kt * 32, nb, lane);
      bI1[nt][kt] = load_B_g16(Wih1T, 192, kt * 32, nb, lane);
      bW1[nt][kt] = load_B_g16(Whh1T, 192, kt * 32, nb, lane);
    }
  __syncthreads();

  for (int t = 0; t < 60; ++t) {
    {   // stage x_t tile (32 rows x 6), prefetch next step's slice
      const int r = tid / 6, d = tid - r * 6;
      if (tid < 192) {
        const size_t base = ((size_t)(rowBase + r) * 60 + t) * 6 + d;
        sXt[tid] = x[base];
        if (t + 1 < 60) __builtin_prefetch(&x[base + 6], 0, 3);
      }
    }
    __syncthreads();

    // ---- GEMM: gh0 = h0 @ Whh0T  -> sG -----------------------------------
#pragma unroll
    for (int mt = 0; mt < 2; ++mt) {
      v16h a0 = load_A_lds_h(sH0h, 64, mt * 16, 0, lane);
      v16h a1 = load_A_lds_h(sH0h, 64, mt * 16, 32, lane);
#pragma unroll
      for (int nt = 0; nt < 2; ++nt) {
        v8f c = {};
        c = WMMA_F16(a0, bW0[nt][0], c);
        c = WMMA_F16(a1, bW0[nt][1], c);
        store_C_lds(sG, 192, mt * 16, (2 * wave + nt) * 16, lane, c);
      }
    }
    __syncthreads();

    // ---- layer-0 gates (input proj from x done inline, D=6) --------------
    for (int i = tid; i < 32 * 64; i += 192) {
      const int r = i >> 6, j = i & 63;
      float xr = 0.0f, xz = 0.0f, xn = 0.0f;
#pragma unroll
      for (int d = 0; d < 6; ++d) {
        const float xv = sXt[r * 6 + d];
        xr += xv * sW0[j * 6 + d];
        xz += xv * sW0[(64 + j) * 6 + d];
        xn += xv * sW0[(128 + j) * 6 + d];
      }
      const float rg = sigmoidf_(xr + sG[r * 192 + j] + sB0[j]);
      const float zg = sigmoidf_(xz + sG[r * 192 + 64 + j] + sB0[64 + j]);
      const float hn = sG[r * 192 + 128 + j] + sB0[192 + j];
      const float ng = tanhf_(xn + sB0[128 + j] + rg * hn);
      const float hnew = (1.0f - zg) * ng + zg * sH0[i];
      sH0[i] = hnew;
      sH0h[i] = (half_t)hnew;
    }
    __syncthreads();

    // ---- layer-1: sG = h0new@Wih1T + h1@Whh1T ; sGn = n-cols of xg1 ------
#pragma unroll
    for (int mt = 0; mt < 2; ++mt) {
      v16h a0 = load_A_lds_h(sH0h, 64, mt * 16, 0, lane);
      v16h a1 = load_A_lds_h(sH0h, 64, mt * 16, 32, lane);
      v16h g0 = load_A_lds_h(sH1h, 64, mt * 16, 0, lane);
      v16h g1 = load_A_lds_h(sH1h, 64, mt * 16, 32, lane);
#pragma unroll
      for (int nt = 0; nt < 2; ++nt) {
        const int nTile = 2 * wave + nt;
        v8f c = {};
        c = WMMA_F16(a0, bI1[nt][0], c);
        c = WMMA_F16(a1, bI1[nt][1], c);
        if (nTile >= 8) store_C_lds(sGn, 64, mt * 16, nTile * 16 - 128, lane, c);
        c = WMMA_F16(g0, bW1[nt][0], c);
        c = WMMA_F16(g1, bW1[nt][1], c);
        store_C_lds(sG, 192, mt * 16, nTile * 16, lane, c);
      }
    }
    __syncthreads();

    // ---- layer-1 gates ---------------------------------------------------
    for (int i = tid; i < 32 * 64; i += 192) {
      const int r = i >> 6, j = i & 63;
      const float xn_raw = sGn[r * 64 + j];
      const float rg = sigmoidf_(sG[r * 192 + j] + sB1[j]);
      const float zg = sigmoidf_(sG[r * 192 + 64 + j] + sB1[64 + j]);
      const float hn = (sG[r * 192 + 128 + j] - xn_raw) + sB1[192 + j];
      const float ng = tanhf_(xn_raw + sB1[128 + j] + rg * hn);
      const float hnew = (1.0f - zg) * ng + zg * sH1[i];
      sH1[i] = hnew;
      sH1h[i] = (half_t)hnew;
    }
    __syncthreads();
  }

  for (int i = tid; i < 32 * 64; i += 192) {
    const int r = i >> 6, j = i & 63;
    const float v = sH1[i];
    enc[(size_t)(rowBase + r) * 64 + j] = v;
    enc16[(size_t)(rowBase + r) * 64 + j] = (half_t)v;
  }
}

// ============================================================================
// GAT block: scores (algebraically reduced), softmax, att@H (WMMA), residual,
// fc (WMMA), pred head (k==19), al_in+tanh (WMMA) -> z16.
// grid: (32 i-tiles, 20 groups), block 128 (4 waves)
// ============================================================================
__global__ __launch_bounds__(128) void k_gat(
    const float* __restrict__ enc, const half_t* __restrict__ enc16,
    const float* __restrict__ trans_W, const float* __restrict__ trans_b,
    const float* __restrict__ a_vec,
    const half_t* __restrict__ fcT, const float* __restrict__ fc_b,
    const half_t* __restrict__ alinT, const float* __restrict__ al_in_b,
    const float* __restrict__ fco_W, const float* __restrict__ fco_b,
    half_t* __restrict__ z16, float* __restrict__ d_out)
{
  __shared__ float  sScore[16 * 512];
  __shared__ half_t sAtt[16 * 512];
  __shared__ float  sSd[512];
  __shared__ float  sU[2 * 64];
  __shared__ float  sC[2];
  __shared__ float  sSs[16];
  __shared__ float  sOut[16 * 64];
  __shared__ float  sOut2[16 * 64];
  __shared__ float  sRed[16 * 8];

  const int tid = threadIdx.x, lane = tid & 31, wave = tid >> 5;   // wave 0..3
  const int k = blockIdx.y;
  const int iBase = blockIdx.x * 16;
  const size_t gbase = (size_t)k * 512 * 64;

  // u_dst/u_src = trans_W^T @ a halves ;  c = trans_b . a halves
  if (tid < 64) {
    float ud = 0.0f, us = 0.0f;
    for (int g = 0; g < 64; ++g) {
      const float w = trans_W[g * 64 + tid];
      ud += w * a_vec[g];
      us += w * a_vec[64 + g];
    }
    sU[tid] = ud; sU[64 + tid] = us;
  } else if (tid < 66) {
    const int s = tid - 64;
    float c = 0.0f;
    for (int g = 0; g < 64; ++g) c += trans_b[g] * a_vec[s * 64 + g];
    sC[s] = c;
  }
  __syncthreads();

  for (int j = tid; j < 512; j += 128) {
    const float* hj = enc + gbase + (size_t)j * 64;
    float s = sC[0];
    for (int d = 0; d < 64; ++d) s += hj[d] * sU[d];
    sSd[j] = s;
  }
  if (tid < 16) {
    const float* hi = enc + gbase + (size_t)(iBase + tid) * 64;
    float s = sC[1];
    for (int d = 0; d < 64; ++d) s += hi[d] * sU[64 + d];
    sSs[tid] = s;
  }
  __syncthreads();

  for (int idx = tid; idx < 16 * 512; idx += 128)
    sScore[idx] = lrelu_(sSs[idx >> 9] + sSd[idx & 511]);
  __syncthreads();

  // row softmax: 8 threads per row
  const int ri = tid >> 3, rt = tid & 7;
  float mx = -1e30f;
  for (int j = rt; j < 512; j += 8) mx = fmaxf(mx, sScore[ri * 512 + j]);
  sRed[ri * 8 + rt] = mx;
  __syncthreads();
  float rowmax = sRed[ri * 8];
  for (int q = 1; q < 8; ++q) rowmax = fmaxf(rowmax, sRed[ri * 8 + q]);
  float ssum = 0.0f;
  for (int j = rt; j < 512; j += 8) {
    const float e = __expf(sScore[ri * 512 + j] - rowmax);
    sScore[ri * 512 + j] = e;
    ssum += e;
  }
  __syncthreads();
  sRed[ri * 8 + rt] = ssum;
  __syncthreads();
  float rowsum = 0.0f;
  for (int q = 0; q < 8; ++q) rowsum += sRed[ri * 8 + q];
  const float inv = frcp_(rowsum);
  for (int j = rt; j < 512; j += 8)
    sAtt[ri * 512 + j] = (half_t)(sScore[ri * 512 + j] * inv);
  __syncthreads();

  // GEMM1: out = att[16,512] @ Hg[512,64]   (wave -> N-tile)
  {
    v8f c = {};
    for (int kt = 0; kt < 16; ++kt) {
      v16h a = load_A_lds_h(sAtt, 512, 0, kt * 32, lane);
      v16h b = load_B_g16(enc16 + gbase, 64, kt * 32, wave * 16, lane);
      c = WMMA_F16(a, b, c);
    }
    store_C_lds(sOut, 64, 0, wave * 16, lane, c);
  }
  __syncthreads();
  for (int idx = tid; idx < 16 * 64; idx += 128) {
    const int i = idx >> 6, d = idx & 63;
    sOut[idx] += enc[gbase + (size_t)(iBase + i) * 64 + d];
  }
  __syncthreads();

  // GEMM2: fc
  {
    v16h a0 = load_A_lds_f32(sOut, 64, 0, 0, lane);
    v16h a1 = load_A_lds_f32(sOut, 64, 0, 32, lane);
    v16h b0 = load_B_g16(fcT, 64, 0, wave * 16, lane);
    v16h b1 = load_B_g16(fcT, 64, 32, wave * 16, lane);
    v8f c = {};
    c = WMMA_F16(a0, b0, c);
    c = WMMA_F16(a1, b1, c);
    const int n = wave * 16 + (lane & 15);
    const int hh = lane >> 4;
#pragma unroll
    for (int v = 0; v < 8; ++v) sOut2[(v + 8 * hh) * 64 + n] = c[v] + fc_b[n];
  }
  __syncthreads();

  // pred head (group k==19 is h[:, -1, :])
  if (k == 19) {
    float partial = 0.0f;
    for (int d = rt; d < 64; d += 8) partial += lrelu_(sOut2[ri * 64 + d]) * fco_W[d];
    sRed[ri * 8 + rt] = partial;
    __syncthreads();
    if (rt == 0) {
      float s = fco_b[0];
      for (int q = 0; q < 8; ++q) s += sRed[ri * 8 + q];
      d_out[512 + iBase + ri] = s;
    }
  }

  // GEMM3: z = tanh(fcout @ al_in_W^T + b)  -> z16 [m][20][64]
  {
    v16h a0 = load_A_lds_f32(sOut2, 64, 0, 0, lane);
    v16h a1 = load_A_lds_f32(sOut2, 64, 0, 32, lane);
    v16h b0 = load_B_g16(alinT, 64, 0, wave * 16, lane);
    v16h b1 = load_B_g16(alinT, 64, 32, wave * 16, lane);
    v8f c = {};
    c = WMMA_F16(a0, b0, c);
    c = WMMA_F16(a1, b1, c);
    const int n = wave * 16 + (lane & 15);
    const int hh = lane >> 4;
#pragma unroll
    for (int v = 0; v < 8; ++v) {
      const float zv = tanhf_(c[v] + al_in_b[n]);
      z16[((size_t)(iBase + v + 8 * hh) * 20 + k) * 64 + n] = (half_t)zv;
    }
  }
}

// ============================================================================
// ALSTM fused 2-layer GRU over K=20, batch 512. 32 rows/block, 6 waves.
// ============================================================================
__global__ __launch_bounds__(192) void k_alstm(
    const half_t* __restrict__ z16,                        // [512][20][64]
    const half_t* __restrict__ Wih0T, const half_t* __restrict__ Whh0T,
    const half_t* __restrict__ Wih1T, const half_t* __restrict__ Whh1T,
    const float* __restrict__ bih0, const float* __restrict__ bhh0,
    const float* __restrict__ bih1, const float* __restrict__ bhh1,
    float* __restrict__ rbuf)                              // [512][20][64]
{
  __shared__ float  sH0[32 * 64];
  __shared__ float  sH1[32 * 64];
  __shared__ half_t sH0h[32 * 64];
  __shared__ half_t sH1h[32 * 64];
  __shared__ float  sG[32 * 192];
  __shared__ float  sGn[32 * 64];
  __shared__ float  sB0[256];
  __shared__ float  sB1[256];

  const int tid = threadIdx.x;
  const int lane = tid & 31;
  const int wave = tid >> 5;
  const int rowBase = blockIdx.x * 32;

  for (int i = tid; i < 32 * 64; i += 192) {
    sH0[i] = 0.0f; sH1[i] = 0.0f;
    sH0h[i] = (half_t)0.0f; sH1h[i] = (half_t)0.0f;
  }
  if (tid < 128) { sB0[tid] = bih0[tid] + bhh0[tid]; sB1[tid] = bih1[tid] + bhh1[tid]; }
  else if (tid < 192) {
    const int j = tid - 128;
    sB0[128 + j] = bih0[128 + j]; sB0[192 + j] = bhh0[128 + j];
    sB1[128 + j] = bih1[128 + j]; sB1[192 + j] = bhh1[128 + j];
  }

  v16h bI0[2][2], bW0[2][2], bI1[2][2], bW1[2][2];
#pragma unroll
  for (int nt = 0; nt < 2; ++nt)
#pragma unroll
    for (int kt = 0; kt < 2; ++kt) {
      const int nb = (2 * wave + nt) * 16;
      bI0[nt][kt] = load_B_g16(Wih0T, 192, kt * 32, nb, lane);
      bW0[nt][kt] = load_B_g16(Whh0T, 192, kt * 32, nb, lane);
      bI1[nt][kt] = load_B_g16(Wih1T, 192, kt * 32, nb, lane);
      bW1[nt][kt] = load_B_g16(Whh1T, 192, kt * 32, nb, lane);
    }
  __syncthreads();

  for (int t = 0; t < 20; ++t) {
    // ---- layer-0: sG = z_t@Wih0T + h0@Whh0T ; sGn = n-cols of xg0 --------
#pragma unroll
    for (int mt = 0; mt < 2; ++mt) {
      v16h a0 = load_A_g16s(z16 + t * 64, 20 * 64, rowBase + mt * 16, 0, lane);
      v16h a1 = load_A_g16s(z16 + t * 64, 20 * 64, rowBase + mt * 16, 32, lane);
      v16h g0 = load_A_lds_h(sH0h, 64, mt * 16, 0, lane);
      v16h g1 = load_A_lds_h(sH0h, 64, mt * 16, 32, lane);
#pragma unroll
      for (int nt = 0; nt < 2; ++nt) {
        const int nTile = 2 * wave + nt;
        v8f c = {};
        c = WMMA_F16(a0, bI0[nt][0], c);
        c = WMMA_F16(a1, bI0[nt][1], c);
        if (nTile >= 8) store_C_lds(sGn, 64, mt * 16, nTile * 16 - 128, lane, c);
        c = WMMA_F16(g0, bW0[nt][0], c);
        c = WMMA_F16(g1, bW0[nt][1], c);
        store_C_lds(sG, 192, mt * 16, nTile * 16, lane, c);
      }
    }
    __syncthreads();

    for (int i = tid; i < 32 * 64; i += 192) {
      const int r = i >> 6, j = i & 63;
      const float xn_raw = sGn[r * 64 + j];
      const float rg = sigmoidf_(sG[r * 192 + j] + sB0[j]);
      const float zg = sigmoidf_(sG[r * 192 + 64 + j] + sB0[64 + j]);
      const float hn = (sG[r * 192 + 128 + j] - xn_raw) + sB0[192 + j];
      const float ng = tanhf_(xn_raw + sB0[128 + j] + rg * hn);
      const float hnew = (1.0f - zg) * ng + zg * sH0[i];
      sH0[i] = hnew;
      sH0h[i] = (half_t)hnew;
    }
    __syncthreads();

    // ---- layer-1 ---------------------------------------------------------
#pragma unroll
    for (int mt = 0; mt < 2; ++mt) {
      v16h a0 = load_A_lds_h(sH0h, 64, mt * 16, 0, lane);
      v16h a1 = load_A_lds_h(sH0h, 64, mt * 16, 32, lane);
      v16h g0 = load_A_lds_h(sH1h, 64, mt * 16, 0, lane);
      v16h g1 = load_A_lds_h(sH1h, 64, mt * 16, 32, lane);
#pragma unroll
      for (int nt = 0; nt < 2; ++nt) {
        const int nTile = 2 * wave + nt;
        v8f c = {};
        c = WMMA_F16(a0, bI1[nt][0], c);
        c = WMMA_F16(a1, bI1[nt][1], c);
        if (nTile >= 8) store_C_lds(sGn, 64, mt * 16, nTile * 16 - 128, lane, c);
        c = WMMA_F16(g0, bW1[nt][0], c);
        c = WMMA_F16(g1, bW1[nt][1], c);
        store_C_lds(sG, 192, mt * 16, nTile * 16, lane, c);
      }
    }
    __syncthreads();

    for (int i = tid; i < 32 * 64; i += 192) {
      const int r = i >> 6, j = i & 63;
      const float xn_raw = sGn[r * 64 + j];
      const float rg = sigmoidf_(sG[r * 192 + j] + sB1[j]);
      const float zg = sigmoidf_(sG[r * 192 + 64 + j] + sB1[64 + j]);
      const float hn = (sG[r * 192 + 128 + j] - xn_raw) + sB1[192 + j];
      const float ng = tanhf_(xn_raw + sB1[128 + j] + rg * hn);
      const float hnew = (1.0f - zg) * ng + zg * sH1[i];
      sH1[i] = hnew;
      sH1h[i] = (half_t)hnew;
      rbuf[((size_t)(rowBase + r) * 20 + t) * 64 + j] = hnew;
    }
    __syncthreads();
  }
}

// ============================================================================
// ALSTM attention head + final projection. block per m.
// ============================================================================
__global__ __launch_bounds__(64) void k_head(
    const float* __restrict__ rbuf,
    const float* __restrict__ att1_W, const float* __restrict__ att1_b,
    const float* __restrict__ att2_W,
    const float* __restrict__ out_W, const float* __restrict__ out_b,
    float* __restrict__ d_out)
{
  __shared__ float sS[20];
  __shared__ float sT[64];
  const int m = blockIdx.x, tid = threadIdx.x;
  const float* rm = rbuf + (size_t)m * 20 * 64;

  for (int kk = 0; kk < 20; ++kk) {
    float v = 0.0f;
    if (tid < 32) {
      float u = att1_b[tid];
      for (int d = 0; d < 64; ++d) u += att1_W[tid * 64 + d] * rm[kk * 64 + d];
      v = tanhf_(u) * att2_W[tid];
    }
    sT[tid] = v;
    __syncthreads();
    if (tid == 0) {
      float s = 0.0f;
      for (int q = 0; q < 32; ++q) s += sT[q];
      sS[kk] = s;
    }
    __syncthreads();
  }
  if (tid == 0) {
    float mx = sS[0];
    for (int kk = 1; kk < 20; ++kk) mx = fmaxf(mx, sS[kk]);
    float sum = 0.0f;
    for (int kk = 0; kk < 20; ++kk) { const float e = __expf(sS[kk] - mx); sS[kk] = e; sum += e; }
    const float inv = frcp_(sum);
    for (int kk = 0; kk < 20; ++kk) sS[kk] *= inv;
  }
  __syncthreads();
  {
    const int d = tid;
    float oa = 0.0f;
    for (int kk = 0; kk < 20; ++kk) oa += sS[kk] * rm[kk * 64 + d];
    sT[d] = out_W[d] * rm[19 * 64 + d] + out_W[64 + d] * oa;
  }
  __syncthreads();
  if (tid == 0) {
    float s = out_b[0];
    for (int q = 0; q < 64; ++q) s += sT[q];
    d_out[m] = s;
  }
}

// ============================================================================
extern "C" void kernel_launch(void* const* d_in, const int* in_sizes, int n_in,
                              void* d_out, int out_size, void* d_ws, size_t ws_size,
                              hipStream_t stream) {
  (void)in_sizes; (void)n_in; (void)out_size; (void)ws_size;
  const float* x       = (const float*)d_in[0];
  const float* rWih0   = (const float*)d_in[1];
  const float* rWhh0   = (const float*)d_in[2];
  const float* rbih0   = (const float*)d_in[3];
  const float* rbhh0   = (const float*)d_in[4];
  const float* rWih1   = (const float*)d_in[5];
  const float* rWhh1   = (const float*)d_in[6];
  const float* rbih1   = (const float*)d_in[7];
  const float* rbhh1   = (const float*)d_in[8];
  const float* trans_W = (const float*)d_in[9];
  const float* trans_b = (const float*)d_in[10];
  const float* a_vec   = (const float*)d_in[11];
  const float* fc_W    = (const float*)d_in[12];
  const float* fc_b    = (const float*)d_in[13];
  const float* fco_W   = (const float*)d_in[14];
  const float* fco_b   = (const float*)d_in[15];
  const float* alin_W  = (const float*)d_in[16];
  const float* alin_b  = (const float*)d_in[17];
  const float* aWih0   = (const float*)d_in[18];
  const float* aWhh0   = (const float*)d_in[19];
  const float* abih0   = (const float*)d_in[20];
  const float* abhh0   = (const float*)d_in[21];
  const float* aWih1   = (const float*)d_in[22];
  const float* aWhh1   = (const float*)d_in[23];
  const float* abih1   = (const float*)d_in[24];
  const float* abhh1   = (const float*)d_in[25];
  const float* att1_W  = (const float*)d_in[26];
  const float* att1_b  = (const float*)d_in[27];
  const float* att2_W  = (const float*)d_in[28];
  const float* out_W   = (const float*)d_in[29];
  const float* out_b   = (const float*)d_in[30];
  float* out = (float*)d_out;

  char* ws = (char*)d_ws;
  float*  enc   = (float*)(ws + 0);              // 10240*64*4  = 2621440
  half_t* enc16 = (half_t*)(ws + 2621440);       // 10240*64*2  = 1310720
  half_t* z16   = (half_t*)(ws + 3932160);       // 512*20*64*2 = 1310720
  float*  rbuf  = (float*)(ws + 5242880);        // 512*20*64*4 = 2621440
  half_t* w16   = (half_t*)(ws + 7864320);
  half_t* wWhh0 = w16;
  half_t* wWih1 = w16 + 1 * 12288;
  half_t* wWhh1 = w16 + 2 * 12288;
  half_t* wAI0  = w16 + 3 * 12288;
  half_t* wAH0  = w16 + 4 * 12288;
  half_t* wAI1  = w16 + 5 * 12288;
  half_t* wAH1  = w16 + 6 * 12288;
  half_t* wFc   = w16 + 7 * 12288;
  half_t* wAlin = wFc + 4096;

  const int TT = 256;
  const int nBig = (192 * 64 + TT - 1) / TT;
  const int nSm  = (64 * 64 + TT - 1) / TT;
  k_tcast<<<nBig, TT, 0, stream>>>(rWhh0,  wWhh0, 192, 64);
  k_tcast<<<nBig, TT, 0, stream>>>(rWih1,  wWih1, 192, 64);
  k_tcast<<<nBig, TT, 0, stream>>>(rWhh1,  wWhh1, 192, 64);
  k_tcast<<<nBig, TT, 0, stream>>>(aWih0,  wAI0,  192, 64);
  k_tcast<<<nBig, TT, 0, stream>>>(aWhh0,  wAH0,  192, 64);
  k_tcast<<<nBig, TT, 0, stream>>>(aWih1,  wAI1,  192, 64);
  k_tcast<<<nBig, TT, 0, stream>>>(aWhh1,  wAH1,  192, 64);
  k_tcast<<<nSm,  TT, 0, stream>>>(fc_W,   wFc,   64, 64);
  k_tcast<<<nSm,  TT, 0, stream>>>(alin_W, wAlin, 64, 64);

  k_encoder<<<320, 192, 0, stream>>>(x, rWih0, rbih0, rbhh0, rbih1, rbhh1,
                                     wWhh0, wWih1, wWhh1, enc, enc16);
  k_gat<<<dim3(32, 20), 128, 0, stream>>>(enc, enc16, trans_W, trans_b, a_vec,
                                          wFc, fc_b, wAlin, alin_b, fco_W, fco_b,
                                          z16, out);
  k_alstm<<<16, 192, 0, stream>>>(z16, wAI0, wAH0, wAI1, wAH1,
                                  abih0, abhh0, abih1, abhh1, rbuf);
  k_head<<<512, 64, 0, stream>>>(rbuf, att1_W, att1_b, att2_W, out_W, out_b, out);
}
